// NMTSelfAttention_39393440039470
// MI455X (gfx1250) — compile-verified, matrix operand order checked by reading, not run on hardware
//
#include <hip/hip_runtime.h>
#include <hip/hip_bf16.h>
#include <math.h>

// ---------------------------------------------------------------------------
// CDNA5 (gfx1250) self-attention, fixed shape B=4, T=2048, H=1024, NH=16,
// NKV=4, HD=64.
//  - all contractions on v_wmma_f32_16x16x32_bf16 (f32 in, bf16 compute, f32 acc)
//  - raw bf16 tile staging via GLOBAL_LOAD_ASYNC_TO_LDS_B128 (ASYNCcnt path)
//  - V workspace stored pre-transposed [b,kvh,d,t] so no in-kernel transpose
//  - all dims compile-time: no integer division, constant-trip staging loops
// ---------------------------------------------------------------------------

typedef __attribute__((ext_vector_type(16))) __bf16         v16bf;
typedef __attribute__((ext_vector_type(16))) unsigned short v16us;
typedef __attribute__((ext_vector_type(8)))  unsigned short v8us;
typedef __attribute__((ext_vector_type(4)))  unsigned short v4us;
typedef __attribute__((ext_vector_type(8)))  float          v8f;

constexpr int Bb  = 4;
constexpr int Td  = 2048;      // sequence length (power of two: >>11, &2047)
constexpr int Hd  = 1024;      // model dim = K of projections
constexpr int NH  = 16;
constexpr int NKV = 4;
constexpr int HD  = 64;
constexpr int Md  = Bb * Td;   // 8192 rows

union FragCast { v16us u; v16bf b; };

static __device__ __forceinline__ unsigned short f2bf_u(float f) {
    unsigned int u = __builtin_bit_cast(unsigned int, f);
    u = u + 0x7FFFu + ((u >> 16) & 1u);          // round-to-nearest-even
    return (unsigned short)(u >> 16);
}

static __device__ __forceinline__ v8f zero8() {
    v8f z;
#pragma unroll
    for (int i = 0; i < 8; ++i) z[i] = 0.0f;
    return z;
}

static __device__ __forceinline__ v8f wmma_bf16(v16us a, v16us b, v8f c) {
    FragCast A; A.u = a;
    FragCast B; B.u = b;
    return __builtin_amdgcn_wmma_f32_16x16x32_bf16(
        false, A.b, false, B.b, (short)0, c, false, false);
}

// Async DMA: 16 bytes global -> LDS, tracked by ASYNCcnt (no VGPR data path).
// LDS byte address = low 32 bits of the flat shared pointer (wave-relative).
static __device__ __forceinline__ void async_copy_b128(void* lds_dst, const void* gsrc) {
    unsigned lds = (unsigned)(uintptr_t)lds_dst;
    asm volatile("global_load_async_to_lds_b128 %0, %1, off"
                 :: "v"(lds), "v"(gsrc) : "memory");
}
static __device__ __forceinline__ void wait_async() {
    asm volatile("s_wait_asynccnt 0x0" ::: "memory");
}

// 16x32 bf16 fragment from row-major storage (stride multiple of 8 elements).
// ISA layout: lane L -> row = L&15, kbase = (L>>4)*8; elems {kb..kb+7, kb+16..kb+23}.
static __device__ __forceinline__ v16us load_frag_rm(const unsigned short* base, int stride) {
    const int lane = threadIdx.x & 31;
    const int row  = lane & 15;
    const int kb   = (lane >> 4) << 3;
    const unsigned short* p = base + (size_t)row * stride + kb;
    v8us lo = *(const v8us*)(p);
    v8us hi = *(const v8us*)(p + 16);
    v16us r;
#pragma unroll
    for (int i = 0; i < 8; ++i) { r[i] = lo[i]; r[8 + i] = hi[i]; }
    return r;
}

// ---------------------------------------------------------------------------
// Kernel 1: projection  bf16_out = x_f32[Md,Hd] @ W_f32[Hd,NCOLS]
// Block: 128 thr (4 waves), tile 128(M) x 64(N), K-step 32; wave owns 32 rows.
// VTRANS==0: out[b, head, t, d]   (Q, K layout)
// VTRANS==1: out[b, head, d, t]   (V layout, pre-transposed for flash P@V)
// ---------------------------------------------------------------------------
template <int NHEADS, int NCOLS, int VTRANS>
__global__ __launch_bounds__(128)
void proj_kernel(const float* __restrict__ A, const float* __restrict__ W,
                 unsigned short* __restrict__ out) {
    __shared__ __align__(16) unsigned short Abf[128][32];   // [m][k] bf16, 8 KB
    __shared__ __align__(16) unsigned short Bt[64][32];     // [n][k] bf16, 4 KB

    const int tid  = threadIdx.x;
    const int wave = tid >> 5;
    const int lane = tid & 31;
    const int m0   = blockIdx.x * 128;
    const int n0   = blockIdx.y * 64;

    v8f acc0[4] = { zero8(), zero8(), zero8(), zero8() };   // rows wave*32+0..15
    v8f acc1[4] = { zero8(), zero8(), zero8(), zero8() };   // rows wave*32+16..31

    for (int k0 = 0; k0 < Hd; k0 += 32) {
        // stage A tile: float4 loads, pack 4 bf16 -> one 8-byte ds store
#pragma unroll
        for (int it = 0; it < 8; ++it) {                     // 1024 float4 / 128 thr
            int j = it * 128 + tid;
            int r = j >> 3, c4 = j & 7;
            float4 f = *(const float4*)(A + (size_t)(m0 + r) * Hd + k0 + c4 * 4);
            v4us p; p[0] = f2bf_u(f.x); p[1] = f2bf_u(f.y);
                    p[2] = f2bf_u(f.z); p[3] = f2bf_u(f.w);
            *(v4us*)(&Abf[r][c4 * 4]) = p;
        }
        // stage W tile transposed: coalesced float4 over n, scatter 4 b16 stores
#pragma unroll
        for (int it = 0; it < 4; ++it) {                     // 512 float4 / 128 thr
            int j = it * 128 + tid;
            int r = j >> 4, c4 = j & 15;                     // r = k row, c4*4 = n
            float4 f = *(const float4*)(W + (size_t)(k0 + r) * NCOLS + n0 + c4 * 4);
            Bt[c4 * 4 + 0][r] = f2bf_u(f.x);
            Bt[c4 * 4 + 1][r] = f2bf_u(f.y);
            Bt[c4 * 4 + 2][r] = f2bf_u(f.z);
            Bt[c4 * 4 + 3][r] = f2bf_u(f.w);
        }
        __syncthreads();

        v16us a0 = load_frag_rm(&Abf[wave * 32 +  0][0], 32);
        v16us a1 = load_frag_rm(&Abf[wave * 32 + 16][0], 32);
#pragma unroll
        for (int t = 0; t < 4; ++t) {
            v16us b = load_frag_rm(&Bt[t * 16][0], 32);
            acc0[t] = wmma_bf16(a0, b, acc0[t]);
            acc1[t] = wmma_bf16(a1, b, acc1[t]);
        }
        __syncthreads();
    }

    // Epilogue: C/D layout -> head-major bf16 (all index math shifts/ands)
    const int half = lane >> 4;
    const int nl   = lane & 15;
#pragma unroll
    for (int g = 0; g < 2; ++g) {
#pragma unroll
        for (int t = 0; t < 4; ++t) {
#pragma unroll
            for (int v = 0; v < 8; ++v) {
                float val = g ? acc1[t][v] : acc0[t][v];
                int mg = m0 + wave * 32 + g * 16 + half * 8 + v;   // row in [Md]
                int ng = n0 + t * 16 + nl;                         // col in [NCOLS]
                int bb = mg >> 11, tt = mg & (Td - 1);
                int hh = ng >> 6,  dd = ng & 63;
                size_t idx = VTRANS
                    ? ((((size_t)bb * NHEADS + hh) * 64 + dd) * Td + tt)   // [b,h,d,t]
                    : ((((size_t)bb * NHEADS + hh) * Td + tt) * 64 + dd);  // [b,h,t,d]
                out[idx] = f2bf_u(val);
            }
        }
    }
}

// ---------------------------------------------------------------------------
// Kernel 2: flash attention, online softmax, causal, GQA (4 q-heads / kv-head).
// grid = (Td/64, NH, Bb); block = 128 (4 waves x 16 q-rows); 32-key blocks.
// K ws layout [b,kvh,t,d]; V ws layout [b,kvh,d,t] (pre-transposed).
// K/V tiles staged with global_load_async_to_lds_b128.
// ---------------------------------------------------------------------------
__global__ __launch_bounds__(128)
void flash_attn_kernel(const unsigned short* __restrict__ Q,
                       const unsigned short* __restrict__ Km,
                       const unsigned short* __restrict__ Vm,
                       unsigned short* __restrict__ O) {
    __shared__ __align__(16) unsigned short Klds[32][64];      // [key][d]
    __shared__ __align__(16) unsigned short Vt[64][32];        // [d][key]
    __shared__ __align__(16) unsigned short Plds[4][16][32];   // per-wave P tile

    const int tid  = threadIdx.x;
    const int wave = tid >> 5;
    const int lane = tid & 31;
    const int half = lane >> 4;
    const int nl   = lane & 15;

    const int h   = blockIdx.y;
    const int b   = blockIdx.z;
    const int q0  = blockIdx.x * 64;
    const int qw  = q0 + wave * 16;
    const int kvh = h >> 2;                      // NH/NKV == 4

    const unsigned short* Qb = Q  + (((size_t)b * NH  + h)   * Td) * 64;
    const unsigned short* Kb = Km + (((size_t)b * NKV + kvh) * Td) * 64;
    const unsigned short* Vb = Vm + (((size_t)b * NKV + kvh) * 64) * Td;  // [d][t]

    // Q fragments in registers: 16 rows x 64 d -> two 16x32 A-frags
    v16us qa0 = load_frag_rm(Qb + (size_t)qw * 64 + 0,  64);
    v16us qa1 = load_frag_rm(Qb + (size_t)qw * 64 + 32, 64);

    v8f o[4] = { zero8(), zero8(), zero8(), zero8() };
    float rmax[8], rsum[8];
#pragma unroll
    for (int v = 0; v < 8; ++v) { rmax[v] = -1.0e30f; rsum[v] = 0.0f; }

    const float scale = 0.125f;                  // 1/sqrt(64)
    const int kmax = q0 + 64;                    // causal bound

    for (int kb0 = 0; kb0 < kmax; kb0 += 32) {
        // Async DMA stage: K tile (32 keys x 64 d) and V tile (64 d x 32 keys),
        // both row-contiguous -> pure b128 copies, zero VALU.
#pragma unroll
        for (int it = 0; it < 2; ++it) {                     // K: 256 chunks
            int i = it * 128 + tid;
            int key = i >> 3, ch = i & 7;
            async_copy_b128(&Klds[key][ch * 8],
                            Kb + (size_t)(kb0 + key) * 64 + ch * 8);
        }
#pragma unroll
        for (int it = 0; it < 2; ++it) {                     // V: 256 chunks
            int i = it * 128 + tid;
            int d = i >> 2, ch = i & 3;
            async_copy_b128(&Vt[d][ch * 8],
                            Vb + (size_t)d * Td + kb0 + ch * 8);
        }
        if (kb0 + 32 < kmax) {                               // prefetch next block
            __builtin_prefetch(Kb + (size_t)(kb0 + 32) * 64 + tid, 0, 1);
            __builtin_prefetch(Vb + (size_t)(kb0 + 32) + tid, 0, 1);
        }
        wait_async();
        __syncthreads();

        // S = Q K^T  (two 16x16 key-tiles, each accumulated over d in 2 steps)
        v8f s0 = zero8(), s1 = zero8();
        {
            v16us b00 = load_frag_rm(&Klds[0][0],       64);
            v16us b01 = load_frag_rm(&Klds[0][0]  + 32, 64);
            v16us b10 = load_frag_rm(&Klds[16][0],      64);
            v16us b11 = load_frag_rm(&Klds[16][0] + 32, 64);
            s0 = wmma_bf16(qa0, b00, s0);
            s0 = wmma_bf16(qa1, b01, s0);
            s1 = wmma_bf16(qa0, b10, s1);
            s1 = wmma_bf16(qa1, b11, s1);
        }

        // scale + causal mask + online softmax (row = half*8+v, col = nl/16+nl)
        float al[8];
#pragma unroll
        for (int v = 0; v < 8; ++v) {
            const int qg = qw + half * 8 + v;
            float x0 = s0[v] * scale;
            float x1 = s1[v] * scale;
            x0 = ((kb0 + nl)      > qg) ? -__builtin_inff() : x0;
            x1 = ((kb0 + 16 + nl) > qg) ? -__builtin_inff() : x1;

            float cm = fmaxf(x0, x1);
#pragma unroll
            for (int m = 1; m < 16; m <<= 1) cm = fmaxf(cm, __shfl_xor(cm, m, 32));
            float nm = fmaxf(rmax[v], cm);                   // finite (rmax init -1e30)

            float alpha = __expf(rmax[v] - nm);
            float p0 = __expf(x0 - nm);
            float p1 = __expf(x1 - nm);
            float ps = p0 + p1;
#pragma unroll
            for (int m = 1; m < 16; m <<= 1) ps += __shfl_xor(ps, m, 32);

            rsum[v] = rsum[v] * alpha + ps;
            rmax[v] = nm;
            al[v]   = alpha;

            Plds[wave][half * 8 + v][nl]      = f2bf_u(p0);  // C-frag -> row-major
            Plds[wave][half * 8 + v][16 + nl] = f2bf_u(p1);
        }
#pragma unroll
        for (int t = 0; t < 4; ++t)
#pragma unroll
            for (int v = 0; v < 8; ++v) o[t][v] *= al[v];

        // O += P @ V  (four 16x16 d-tiles)
        v16us pa = load_frag_rm(&Plds[wave][0][0], 32);
#pragma unroll
        for (int t = 0; t < 4; ++t) {
            v16us bv = load_frag_rm(&Vt[t * 16][0], 32);
            o[t] = wmma_bf16(pa, bv, o[t]);
        }
        __syncthreads();   // before next block's async copies overwrite tiles
    }

    // normalize + store to attn workspace [b*Td + t, h*64 + d] (bf16)
#pragma unroll
    for (int v = 0; v < 8; ++v) {
        float inv = (rsum[v] > 0.0f) ? (1.0f / rsum[v]) : 0.0f;
        int qg = qw + half * 8 + v;
#pragma unroll
        for (int t = 0; t < 4; ++t) {
            int dg = t * 16 + nl;
            O[((size_t)b * Td + qg) * ((size_t)NH * 64) + (size_t)h * 64 + dg] =
                f2bf_u(o[t][v] * inv);
        }
    }
}

// ---------------------------------------------------------------------------
// Kernel 3: output projection  out_f32[Md,Hd] = attn_bf16[Md,1024] @ wo_f32
// A tile staged with async DMA (already bf16); W converted through LDS.
// ---------------------------------------------------------------------------
__global__ __launch_bounds__(128)
void out_proj_kernel(const unsigned short* __restrict__ A, const float* __restrict__ W,
                     float* __restrict__ out) {
    constexpr int K = NH * HD;    // 1024
    constexpr int N = Hd;         // 1024
    __shared__ __align__(16) unsigned short Abf[128][32];
    __shared__ __align__(16) unsigned short Bt[64][32];

    const int tid  = threadIdx.x;
    const int wave = tid >> 5;
    const int lane = tid & 31;
    const int m0   = blockIdx.x * 128;
    const int n0   = blockIdx.y * 64;

    v8f acc0[4] = { zero8(), zero8(), zero8(), zero8() };
    v8f acc1[4] = { zero8(), zero8(), zero8(), zero8() };

    for (int k0 = 0; k0 < K; k0 += 32) {
        // A tile: raw bf16 copy -> async DMA b128 (128 rows x 64 B)
#pragma unroll
        for (int it = 0; it < 4; ++it) {                     // 512 chunks
            int j = it * 128 + tid;
            int r = j >> 2, ch = j & 3;
            async_copy_b128(&Abf[r][ch * 8], A + (size_t)(m0 + r) * K + k0 + ch * 8);
        }
        // W tile: convert + transpose
#pragma unroll
        for (int it = 0; it < 4; ++it) {
            int j = it * 128 + tid;
            int r = j >> 4, c4 = j & 15;
            float4 f = *(const float4*)(W + (size_t)(k0 + r) * N + n0 + c4 * 4);
            Bt[c4 * 4 + 0][r] = f2bf_u(f.x);
            Bt[c4 * 4 + 1][r] = f2bf_u(f.y);
            Bt[c4 * 4 + 2][r] = f2bf_u(f.z);
            Bt[c4 * 4 + 3][r] = f2bf_u(f.w);
        }
        wait_async();
        __syncthreads();

        v16us a0 = load_frag_rm(&Abf[wave * 32 +  0][0], 32);
        v16us a1 = load_frag_rm(&Abf[wave * 32 + 16][0], 32);
#pragma unroll
        for (int t = 0; t < 4; ++t) {
            v16us b = load_frag_rm(&Bt[t * 16][0], 32);
            acc0[t] = wmma_bf16(a0, b, acc0[t]);
            acc1[t] = wmma_bf16(a1, b, acc1[t]);
        }
        __syncthreads();
    }

    const int half = lane >> 4;
    const int nl   = lane & 15;
#pragma unroll
    for (int g = 0; g < 2; ++g) {
#pragma unroll
        for (int t = 0; t < 4; ++t) {
#pragma unroll
            for (int v = 0; v < 8; ++v) {
                int mg = m0 + wave * 32 + g * 16 + half * 8 + v;
                int ng = n0 + t * 16 + nl;
                out[(size_t)mg * N + ng] = g ? acc1[t][v] : acc0[t][v];
            }
        }
    }
}

// ---------------------------------------------------------------------------
// Host launcher. Workspace (bf16 elements):
//   Qws [B,NH,T,64] | Kws [B,NKV,T,64] | Vws [B,NKV,64,T] | Aws [B*T, NH*64]
// ---------------------------------------------------------------------------
extern "C" void kernel_launch(void* const* d_in, const int* in_sizes, int n_in,
                              void* d_out, int out_size, void* d_ws, size_t ws_size,
                              hipStream_t stream) {
    (void)in_sizes; (void)n_in; (void)out_size; (void)ws_size;

    const float* x  = (const float*)d_in[0];
    const float* wq = (const float*)d_in[1];
    const float* wk = (const float*)d_in[2];
    const float* wv = (const float*)d_in[3];
    const float* wo = (const float*)d_in[4];
    float* out = (float*)d_out;

    unsigned short* Qws = (unsigned short*)d_ws;
    unsigned short* Kws = Qws + (size_t)Bb * NH  * Td * HD;
    unsigned short* Vws = Kws + (size_t)Bb * NKV * Td * HD;
    unsigned short* Aws = Vws + (size_t)Bb * NKV * Td * HD;

    dim3 blk(128, 1, 1);

    proj_kernel<NH, NH * HD, 0><<<dim3(Md / 128, (NH * HD) / 64), blk, 0, stream>>>(
        x, wq, Qws);
    proj_kernel<NKV, NKV * HD, 0><<<dim3(Md / 128, (NKV * HD) / 64), blk, 0, stream>>>(
        x, wk, Kws);
    proj_kernel<NKV, NKV * HD, 1><<<dim3(Md / 128, (NKV * HD) / 64), blk, 0, stream>>>(
        x, wv, Vws);

    flash_attn_kernel<<<dim3(Td / 64, NH, Bb), blk, 0, stream>>>(Qws, Kws, Vws, Aws);

    out_proj_kernel<<<dim3(Md / 128, Hd / 64), blk, 0, stream>>>(Aws, wo, out);
}
